// CRFDiceHead_6932077216417
// MI455X (gfx1250) — matrix operand order repaired and checked by reading, not compiled, not run on hardware
//
#include <hip/hip_runtime.h>
#include <hip/hip_bf16.h>

#define Bq 128
#define Tq 4096
#define Nq 32

typedef __attribute__((ext_vector_type(16))) _Float16 v16h;
typedef __attribute__((ext_vector_type(8)))  _Float16 v8h;
typedef __attribute__((ext_vector_type(2)))  _Float16 h2;
typedef __attribute__((ext_vector_type(8)))  float    v8f;

#define LOG2E 1.44269504088896340736f

// ---- raw hardware ops (no guard code) ----
__device__ __forceinline__ float fexp2(float x) {
#if __has_builtin(__builtin_amdgcn_exp2f)
    return __builtin_amdgcn_exp2f(x);
#else
    return exp2f(x);
#endif
}
__device__ __forceinline__ float flog2(float x) {
#if __has_builtin(__builtin_amdgcn_logf)
    return __builtin_amdgcn_logf(x);
#else
    return log2f(x);
#endif
}
__device__ __forceinline__ int fexpo(float x) {      // v_frexp_exp_i32_f32
#if __has_builtin(__builtin_amdgcn_frexp_expf)
    return __builtin_amdgcn_frexp_expf(x);
#else
    int e; (void)frexpf(x, &e); return e;
#endif
}
__device__ __forceinline__ float fldx(float x, int e) {  // v_ldexp_f32
#if __has_builtin(__builtin_amdgcn_ldexpf)
    return __builtin_amdgcn_ldexpf(x, e);
#else
    return ldexpf(x, e);
#endif
}

// ---- VALU-latency xor-shuffle within 16-lane rows via v_permlane16_b32 ----
__device__ __forceinline__ float pl16(float v, unsigned lo, unsigned hi, int xm) {
#if __has_builtin(__builtin_amdgcn_permlane16)
    return __int_as_float(__builtin_amdgcn_permlane16(
        __float_as_int(v), __float_as_int(v), (int)lo, (int)hi, false, false));
#else
    return __shfl_xor(v, xm, 32);
#endif
}
__device__ __forceinline__ float rowmax16(float v) {
    v = fmaxf(v, pl16(v, 0x67452301u, 0xEFCDAB89u, 1));
    v = fmaxf(v, pl16(v, 0x54761032u, 0xDCFE98BAu, 2));
    v = fmaxf(v, pl16(v, 0x32107654u, 0xBA98FEDCu, 4));
    v = fmaxf(v, pl16(v, 0xFEDCBA98u, 0x76543210u, 8));
    return v;
}
__device__ __forceinline__ float rowsum16(float v) {
    v += pl16(v, 0x67452301u, 0xEFCDAB89u, 1);
    v += pl16(v, 0x54761032u, 0xDCFE98BAu, 2);
    v += pl16(v, 0x32107654u, 0xBA98FEDCu, 4);
    v += pl16(v, 0xFEDCBA98u, 0x76543210u, 8);
    return v;
}

// ---------------------------------------------------------------------------
// Kernel 1: CRF scan with state kept as normalized LINEAR vector + exponent
// offset:  alpha_linear = 2^ofs * La,  max(La) in [0.5,1) (fwd invariant).
// Per step the serial chain is only: (mul) -> rowmax(permlane16) -> frexp ->
// ldexp -> cvt_pk -> LDS -> v_wmma -> (mul) -> select.  The 16 exp2's for
// 2^(pot*log2e) depend only on the async-prefetched pot tile and run in the
// WMMA shadow; no logs in the loop at all.
// ---------------------------------------------------------------------------
template<bool FWD>
__device__ __forceinline__ void crf_scan_dir(const float* __restrict__ pots,
                                             const int*   __restrict__ lengths,
                                             const float* __restrict__ trans,
                                             float*       __restrict__ outbuf,  // linear trace
                                             float*       __restrict__ ofsbuf,  // [B][T] exponent offsets
                                             float*       __restrict__ logZ,    // base-2
                                             float*       __restrict__ acc,
                                             int blk)
{
    __shared__ __align__(64) h2    lds_a[16][16];       // normalized linear f16 vectors
    __shared__ __align__(16) float lds_pot[2][16][32];  // double-buffered pot tiles

    const int l  = threadIdx.x;
    const int hi = l >> 4;
    const int c  = l & 15;
    const int b0 = blk * 16;

    int lenv[8];
#pragma unroll
    for (int r = 0; r < 8; ++r) lenv[r] = lengths[b0 + r + 8 * hi];

    if (FWD && l < 16) { acc[b0 + l] = 0.f; acc[Bq + b0 + l] = 0.f; }

    // resident B tiles: exp(trans) = 2^(trans*log2e); tile tau = columns 2c+tau
    v16h Bt0, Bt1;
#pragma unroll
    for (int tau = 0; tau < 2; ++tau) {
        v16h bt;
        const int col = 2 * c + tau;
#pragma unroll
        for (int h = 0; h < 16; ++h) {
            const int p = h + 16 * hi;   // K index == source state
            const float tv = FWD ? trans[p * Nq + col] : trans[col * Nq + p];
            bt[h] = (_Float16)fexp2(tv * LOG2E);
        }
        if (tau == 0) Bt0 = bt; else Bt1 = bt;
    }

    const float* prow[8];
    float*       orow[8];
    float*       frow[8];
#pragma unroll
    for (int r = 0; r < 8; ++r) {
        const size_t bidx = (size_t)(b0 + r + 8 * hi);
        prow[r] = pots   + bidx * Tq * Nq + 2 * c;     // states 2c, 2c+1
        orow[r] = outbuf + bidx * Tq * Nq + 2 * c;
        frow[r] = ofsbuf + bidx * Tq;
    }

    // ---- init state ----
    float La[2][8];
    int   ofs[8];
    if (FWD) {
#pragma unroll
        for (int r = 0; r < 8; ++r) {
            const float2 p0 = *(const float2*)&prow[r][0];
            const float x0 = p0.x * LOG2E, x1 = p0.y * LOG2E;
            const float m2 = rowmax16(fmaxf(x0, x1));
            const int   e0 = (int)ceilf(m2);
            La[0][r] = fexp2(x0 - (float)e0);
            La[1][r] = fexp2(x1 - (float)e0);
            ofs[r]   = e0;
            *(float2*)&orow[r][0] = make_float2(La[0][r], La[1][r]);
        }
        if (c == 0) {
#pragma unroll
            for (int r = 0; r < 8; ++r) frow[r][0] = (float)ofs[r];
        }
    } else {
#pragma unroll
        for (int r = 0; r < 8; ++r) {
            La[0][r] = 1.f; La[1][r] = 1.f; ofs[r] = 0;
            *(float2*)&orow[r][(size_t)(Tq - 1) * Nq] = make_float2(1.f, 1.f);
        }
        if (c == 0) {
#pragma unroll
            for (int r = 0; r < 8; ++r) frow[r][Tq - 1] = 0.f;
        }
    }

    // async global->LDS prefetch of pots[b0..b0+15][t][:] (2KB = 4 x b128/lane)
    auto prefetch_pot = [&](int t, int buf) {
#pragma unroll
        for (int k = 0; k < 4; ++k) {
            const int idx = k * 32 + l;
            const int m   = idx >> 3;
            const int ch  = idx & 7;
            const float* g = pots + ((size_t)(b0 + m) * Tq + t) * Nq + ch * 4;
            unsigned           lo = (unsigned)(size_t)&lds_pot[buf][m][ch * 4];
            unsigned long long ga = (unsigned long long)(size_t)g;
            asm volatile("global_load_async_to_lds_b128 %0, %1, off"
                         :: "v"(lo), "v"(ga) : "memory");
        }
    };

    const int pt0 = FWD ? 1 : (Tq - 1);
    prefetch_pot(pt0, pt0 & 1);

    const v8f czero = {};

    for (int step = 1; step <= Tq - 1; ++step) {
        const int t   = FWD ? step : (Tq - 1 - step);  // fwd 1..T-1 ; bwd T-2..0
        const int pt  = FWD ? t : (t + 1);
        const int buf = pt & 1;

        asm volatile("s_wait_asynccnt 0" ::: "memory");
        const int nt = FWD ? (t + 1) : t;
        if (step < Tq - 1) prefetch_pot(nt, nt & 1);

        // pexp = 2^(pot*log2e): depends only on the pot tile -> off the chain
        float pe0[8], pe1[8];
#pragma unroll
        for (int r = 0; r < 8; ++r) {
            const float2 pp = *(const float2*)&lds_pot[buf][r + 8 * hi][2 * c];
            pe0[r] = fexp2(pp.x * LOG2E);
            pe1[r] = fexp2(pp.y * LOG2E);
        }

        // WMMA operand values (bwd folds pot in before the matmul + normalizes)
        float w0[8], w1[8];
        int   ev[8];
#pragma unroll
        for (int r = 0; r < 8; ++r) {
            if (FWD) {                       // La already normalized
                w0[r] = La[0][r]; w1[r] = La[1][r];
            } else {
                float v0 = La[0][r] * pe0[r];
                float v1 = La[1][r] * pe1[r];
                const float m = rowmax16(fmaxf(v0, v1));
                ev[r] = fexpo(m);
                w0[r] = fldx(v0, -ev[r]);
                w1[r] = fldx(v1, -ev[r]);
            }
        }

        // pack f16 pairs -> natural-order LDS rows
#pragma unroll
        for (int r = 0; r < 8; ++r) {
            h2 pk;
            pk.x = (_Float16)w0[r];
            pk.y = (_Float16)w1[r];
            lds_a[r + 8 * hi][c] = pk;
        }

        // A-operand: lane<16 -> K 0..15 of row c ; lane>=16 -> 16..31
        const int e0 = hi ? 8 : 0;
        v8h alo = *(const v8h*)&lds_a[c][e0];
        v8h ahi = *(const v8h*)&lds_a[c][e0 + 4];
        v16h A;
#pragma unroll
        for (int h = 0; h < 8; ++h) { A[h] = alo[h]; A[h + 8] = ahi[h]; }

        v8f D0 = __builtin_amdgcn_wmma_f32_16x16x32_f16(false, A, false, Bt0,
                                                        (short)0, czero, false, false);
        v8f D1 = __builtin_amdgcn_wmma_f32_16x16x32_f16(false, A, false, Bt1,
                                                        (short)0, czero, false, false);

#pragma unroll
        for (int r = 0; r < 8; ++r) {
            if (FWD) {
                float r0 = D0[r] * pe0[r];
                float r1 = D1[r] * pe1[r];
                const float m = rowmax16(fmaxf(r0, r1));
                const int  ee = fexpo(m);
                r0 = fldx(r0, -ee);
                r1 = fldx(r1, -ee);
                const bool valid = t < lenv[r];
                La[0][r] = valid ? r0 : La[0][r];
                La[1][r] = valid ? r1 : La[1][r];
                ofs[r]   = valid ? (ofs[r] + ee) : ofs[r];
            } else {
                const bool valid = (t + 1) < lenv[r];
                La[0][r] = valid ? D0[r] : 1.f;
                La[1][r] = valid ? D1[r] : 1.f;
                ofs[r]   = valid ? (ofs[r] + ev[r]) : 0;
            }
            *(float2*)&orow[r][(size_t)t * Nq] = make_float2(La[0][r], La[1][r]);
        }
        if (c == 0) {
#pragma unroll
            for (int r = 0; r < 8; ++r) frow[r][t] = (float)ofs[r];
        }
    }

    if (FWD) {
        // logZ2[b] = ofs + log2(sum of final linear alpha)
#pragma unroll
        for (int r = 0; r < 8; ++r) {
            const float s = rowsum16(La[0][r] + La[1][r]);
            if (c == 0) logZ[b0 + r + 8 * hi] = (float)ofs[r] + flog2(s);
        }
    }
}

__global__ __launch_bounds__(32)
void crf_scan_kernel(const float* __restrict__ pots,
                     const int*   __restrict__ lengths,
                     const float* __restrict__ trans,
                     float* __restrict__ alphas, float* __restrict__ ofsA,
                     float* __restrict__ betas,  float* __restrict__ ofsB,
                     float* __restrict__ logZ,
                     float* __restrict__ acc)
{
    if (blockIdx.x < 8)
        crf_scan_dir<true >(pots, lengths, trans, alphas, ofsA, logZ, acc, blockIdx.x);
    else
        crf_scan_dir<false>(pots, lengths, trans, betas,  ofsB, logZ, acc, blockIdx.x - 8);
}

// ---------------------------------------------------------------------------
// Kernel 2: marginals + dice partial sums.  One exp2 per (b,t):
//   p_n = LA_n * LB_n * 2^(ofsA + ofsB - logZ2)
// ---------------------------------------------------------------------------
__global__ __launch_bounds__(256)
void crf_combine_kernel(const float* __restrict__ alphas,
                        const float* __restrict__ ofsA,
                        const float* __restrict__ betas,
                        const float* __restrict__ ofsB,
                        const float* __restrict__ logZ,
                        const int*   __restrict__ y_true,
                        const int*   __restrict__ lengths,
                        float*       __restrict__ acc)
{
    const int b = blockIdx.x >> 4;
    const int t = ((blockIdx.x & 15) << 8) + threadIdx.x;

    float inter = 0.f, psum = 0.f;
    const int len = lengths[b];
    if (t < len) {
        const size_t bt  = (size_t)b * Tq + t;
        const size_t off = bt * Nq;
        const float  f   = fexp2(ofsA[bt] + ofsB[bt] - logZ[b]);
        const int    y   = y_true[bt];
        const float4* av = (const float4*)(alphas + off);
        const float4* bv = (const float4*)(betas  + off);
#pragma unroll
        for (int q = 0; q < 8; ++q) {
            const float4 a4 = av[q];
            const float4 b4 = bv[q];
            float p[4];
            p[0] = a4.x * b4.x * f;
            p[1] = a4.y * b4.y * f;
            p[2] = a4.z * b4.z * f;
            p[3] = a4.w * b4.w * f;
#pragma unroll
            for (int j = 0; j < 4; ++j) {
                psum += p[j];
                if (4 * q + j == y) inter = p[j];
            }
        }
    }
#pragma unroll
    for (int o = 16; o > 0; o >>= 1) {
        inter += __shfl_down(inter, o);
        psum  += __shfl_down(psum,  o);
    }
    if ((threadIdx.x & 31) == 0) {
        atomicAdd(&acc[b],      inter);
        atomicAdd(&acc[Bq + b], psum);
    }
}

// ---------------------------------------------------------------------------
__global__ void crf_finalize_kernel(const float* __restrict__ acc,
                                    const int*   __restrict__ lengths,
                                    float*       __restrict__ out)
{
    const int b = blockIdx.x * blockDim.x + threadIdx.x;
    if (b < Bq) {
        const float inter = acc[b];
        const float psum  = acc[Bq + b];
        const float ysum  = (float)lengths[b];
        const float dice  = (2.f * inter + 1.f) / (ysum + psum + 1.f);
        out[b] = 1.f - dice;
    }
}

// ---------------------------------------------------------------------------
extern "C" void kernel_launch(void* const* d_in, const int* in_sizes, int n_in,
                              void* d_out, int out_size, void* d_ws, size_t ws_size,
                              hipStream_t stream)
{
    const float* pots    = (const float*)d_in[0];   // [B,T,N] f32
    const int*   y_true  = (const int*)  d_in[1];   // [B,T]   i32
    const int*   lengths = (const int*)  d_in[2];   // [B]     i32
    const float* trans   = (const float*)d_in[3];   // [N,N]   f32
    float*       out     = (float*)d_out;           // [B]     f32

    float* alphas = (float*)d_ws;                        // linear traces
    float* betas  = alphas + (size_t)Bq * Tq * Nq;
    float* ofsA   = betas  + (size_t)Bq * Tq * Nq;       // [B][T]
    float* ofsB   = ofsA   + (size_t)Bq * Tq;
    float* logZ   = ofsB   + (size_t)Bq * Tq;            // base-2
    float* acc    = logZ + Bq;                           // [2*Bq]

    crf_scan_kernel<<<16, 32, 0, stream>>>(pots, lengths, trans,
                                           alphas, ofsA, betas, ofsB, logZ, acc);
    crf_combine_kernel<<<Bq * 16, 256, 0, stream>>>(alphas, ofsA, betas, ofsB,
                                                    logZ, y_true, lengths, acc);
    crf_finalize_kernel<<<1, 128, 0, stream>>>(acc, lengths, out);
}
